// EncLayer_54357106098666
// MI455X (gfx1250) — compile-verified, matrix operand order checked by reading, not run on hardware
//
#include <hip/hip_runtime.h>

typedef unsigned short u16;
typedef __attribute__((ext_vector_type(16))) __bf16 v16bf;
typedef __attribute__((ext_vector_type(8)))  float  v8f;

#define H 128

#if defined(__HIP_DEVICE_COMPILE__) && __has_builtin(__builtin_amdgcn_global_load_async_to_lds_b128)
#define HAVE_ASYNC_LDS 1
#else
#define HAVE_ASYNC_LDS 0
#endif

#if HAVE_ASYNC_LDS
// Builtin prototype (from hipcc diagnostic): first param is
// "__attribute__((vector_size(16))) int __device__ *" i.e. int4* in AS1.
typedef int vi4 __attribute__((vector_size(16)));
typedef vi4 __attribute__((address_space(1))) gvi4;
typedef vi4 __attribute__((address_space(3))) lvi4;
#endif

__device__ __forceinline__ u16 f2bf(float f){
  union { float f; unsigned u; } a; a.f = f;
  unsigned r = a.u + 0x7FFFu + ((a.u >> 16) & 1u);  // round-to-nearest-even
  return (u16)(r >> 16);
}

// Branch-free gelu: x * sigmoid(1.5958*(x + 0.044715 x^3)) == 0.5x(1+tanh(...))
// One v_exp_f32 + one v_rcp_f32 (TRANS ops, co-execute with WMMA), no EXEC branching.
__device__ __forceinline__ float gelu_f(float x){
  float u = 1.5957691216f * (x + 0.044715f * x * x * x);
  return x * __builtin_amdgcn_rcpf(1.f + __expf(-u));
}

union FragU { v16bf v; unsigned u[8]; };

// 16x32 bf16 WMMA operand fragment, CDNA5 layout:
//   VGPR j<4 : k = 2j   + 8h   (k 0..15 region)   h = lane>>4
//   VGPR j>=4: k = 16+2(j-4)+8h (k 16..31 region)
__device__ __forceinline__ v16bf load_frag(const u16* p, int h){
  FragU f;
#pragma unroll
  for (int j = 0; j < 8; ++j){
    int k = (j < 4) ? (2*j + 8*h) : (16 + 2*(j-4) + 8*h);
    f.u[j] = *(const unsigned*)(p + k);
  }
  return f.v;
}

__device__ __forceinline__ v8f wmma_bf16(v16bf a, v16bf b, v8f c){
  return __builtin_amdgcn_wmma_f32_16x16x32_bf16(false, a, false, b, (short)0, c,
                                                 false, false);
}

// 16-byte global->LDS copy: async path (ASYNCcnt-tracked) when available,
// else a plain vector copy.
__device__ __forceinline__ void cp16(void* lds_dst, const void* gsrc){
#if HAVE_ASYNC_LDS
  __builtin_amdgcn_global_load_async_to_lds_b128(
      (gvi4*)(void*)gsrc, (lvi4*)lds_dst, 0, 0);
#else
  *(uint4*)lds_dst = *(const uint4*)gsrc;
#endif
}

__device__ __forceinline__ void cp16_wait(){
#if HAVE_ASYNC_LDS
#if __has_builtin(__builtin_amdgcn_s_wait_asynccnt)
  __builtin_amdgcn_s_wait_asynccnt(0);
#else
  asm volatile("s_wait_asynccnt 0x0" ::: "memory");
#endif
#endif
}

__device__ __forceinline__ float block_sum(float v, float* red){
  int t = threadIdx.x;
  red[t] = v;
  __syncthreads();
#pragma unroll
  for (int s = 128; s > 0; s >>= 1){
    if (t < s) red[t] += red[t + s];
    __syncthreads();
  }
  float r = red[0];
  __syncthreads();
  return r;
}

// Weight W [Kd x 128] fp32 -> bf16 transposed Wt [128 x Kd] (contiguous B-frags).
__global__ __launch_bounds__(256) void wconv_kernel(const float* __restrict__ W,
                                                    u16* __restrict__ Wt, int Kd){
  int i = blockIdx.x * 256 + threadIdx.x;
  if (i < Kd * H){
    int n = i / Kd, k = i - n * Kd;
    Wt[(size_t)n * Kd + k] = f2bf(W[(size_t)k * H + n]);
  }
}

// Bulk fp32 -> bf16 conversion, 4 elements/thread (all our sizes are /4).
__global__ __launch_bounds__(256) void bconv_kernel(const float* __restrict__ x,
                                                    u16* __restrict__ y, int n4){
  int i = blockIdx.x * 256 + threadIdx.x;
  if (i < n4){
    float4 v = ((const float4*)x)[i];
    ushort4 o;
    o.x = f2bf(v.x); o.y = f2bf(v.y); o.z = f2bf(v.z); o.w = f2bf(v.w);
    ((ushort4*)y)[i] = o;
  }
}

// One block per node; 8 waves, wave w owns output cols [16w,16w+16).
// Async-stages h_EV [48 x 640] bf16 into LDS, 640->128->128->128 WMMA chain,
// masked K-sum, LN + FFN + LN fused. Writes fp32 h_V and a bf16 copy.
__global__ __launch_bounds__(256) void node_kernel(
  const float* __restrict__ hV,
  const u16* __restrict__ hVbf, const u16* __restrict__ hVabf,
  const u16* __restrict__ hEbf,
  const u16* __restrict__ W1t, const float* __restrict__ b1,
  const u16* __restrict__ W2t, const float* __restrict__ b2,
  const u16* __restrict__ W3t, const float* __restrict__ b3,
  const float* __restrict__ Win, const float* __restrict__ bin,
  const float* __restrict__ Wout, const float* __restrict__ bout,
  const float* __restrict__ g1, const float* __restrict__ be1,
  const float* __restrict__ g2, const float* __restrict__ be2,
  const float* __restrict__ maskV, const float* __restrict__ maskA,
  const int* __restrict__ Eidx, float* __restrict__ outV,
  u16* __restrict__ hVnbf, int Nn, int Kk)
{
  __shared__ u16   As[48*648];   // padded ld=648 (row stride 1296B = 81*16B)
  __shared__ u16   A2s[48*132];  // padded ld=132
  __shared__ float dhs[H];
  __shared__ float masks[48];
  __shared__ int   idxs[48];
  __shared__ float red[256];
  __shared__ float hvln[H];
  __shared__ float hid[4*H];

  const int node = blockIdx.x;       // b*N + n
  const int bidx = node / Nn;
  const int tid  = threadIdx.x;
  const int lane = tid & 31;
  const int w    = tid >> 5;
  const int hh   = lane >> 4;
  const int l16  = lane & 15;
  const int ncol = (w << 4) + l16;

  const float* hVrow   = hV   + (size_t)node * H;
  const u16*   hVbfrow = hVbf + (size_t)node * H;
  const u16*   hEbfrow = hEbf + (size_t)node * Kk * H;

  if (tid < 48){
    idxs[tid]  = Eidx[(size_t)node*Kk + tid];
    masks[tid] = maskA[(size_t)node*Kk + tid];
  }
  if (tid < H) dhs[tid] = 0.f;
  __syncthreads();

  // ---- async-stage A = [h_V | h_E | gather(h_V) | h_E | gather(h_V_atom)] ----
  // 48 rows x 5 segs x 16 chunks of 16B
  for (int q = tid; q < 48*80; q += 256){
    int r   = q / 80;
    int s   = q - r*80;
    int seg = s >> 4;
    int o8  = (s & 15) << 3;          // element offset within segment
    const u16* src;
    if (seg == 0)      src = hVbfrow + o8;
    else if (seg == 1) src = hEbfrow + r*H + o8;
    else if (seg == 2) src = hVbf  + ((size_t)bidx*Nn + idxs[r])*H + o8;
    else if (seg == 3) src = hEbfrow + r*H + o8;
    else               src = hVabf + ((size_t)bidx*Nn + idxs[r])*H + o8;
    cp16(&As[r*648 + seg*128 + o8], src);
  }
  cp16_wait();
  __syncthreads();

  // ---- GEMM1: [48x640] @ [640x128] ----
  v8f acc0 = {}, acc1 = {}, acc2 = {};
  for (int kt = 0; kt < 20; ++kt){
    v16bf bf = load_frag(W1t + (size_t)ncol*640 + kt*32, hh);
    v16bf a0 = load_frag(&As[(l16     )*648 + kt*32], hh);
    v16bf a1 = load_frag(&As[(l16 + 16)*648 + kt*32], hh);
    v16bf a2 = load_frag(&As[(l16 + 32)*648 + kt*32], hh);
    acc0 = wmma_bf16(a0, bf, acc0);
    acc1 = wmma_bf16(a1, bf, acc1);
    acc2 = wmma_bf16(a2, bf, acc2);
  }
  {
    float bb = b1[ncol];
#pragma unroll
    for (int vv = 0; vv < 8; ++vv){
      int rr = vv + 8*hh;
      A2s[(rr     )*132 + ncol] = f2bf(gelu_f(acc0[vv] + bb));
      A2s[(rr + 16)*132 + ncol] = f2bf(gelu_f(acc1[vv] + bb));
      A2s[(rr + 32)*132 + ncol] = f2bf(gelu_f(acc2[vv] + bb));
    }
  }
  __syncthreads();

  // ---- GEMM2: [48x128] @ [128x128] ----
  v8f c0 = {}, c1 = {}, c2 = {};
  for (int kt = 0; kt < 4; ++kt){
    v16bf bf = load_frag(W2t + (size_t)ncol*128 + kt*32, hh);
    v16bf a0 = load_frag(&A2s[(l16     )*132 + kt*32], hh);
    v16bf a1 = load_frag(&A2s[(l16 + 16)*132 + kt*32], hh);
    v16bf a2 = load_frag(&A2s[(l16 + 32)*132 + kt*32], hh);
    c0 = wmma_bf16(a0, bf, c0);
    c1 = wmma_bf16(a1, bf, c1);
    c2 = wmma_bf16(a2, bf, c2);
  }
  __syncthreads();   // all reads of A2s done before overwrite
  {
    float bb = b2[ncol];
#pragma unroll
    for (int vv = 0; vv < 8; ++vv){
      int rr = vv + 8*hh;
      A2s[(rr     )*132 + ncol] = f2bf(gelu_f(c0[vv] + bb));
      A2s[(rr + 16)*132 + ncol] = f2bf(gelu_f(c1[vv] + bb));
      A2s[(rr + 32)*132 + ncol] = f2bf(gelu_f(c2[vv] + bb));
    }
  }
  __syncthreads();

  // ---- GEMM3: [48x128] @ [128x128], masked sum over K ----
  v8f d0 = {}, d1 = {}, d2 = {};
  for (int kt = 0; kt < 4; ++kt){
    v16bf bf = load_frag(W3t + (size_t)ncol*128 + kt*32, hh);
    v16bf a0 = load_frag(&A2s[(l16     )*132 + kt*32], hh);
    v16bf a1 = load_frag(&A2s[(l16 + 16)*132 + kt*32], hh);
    v16bf a2 = load_frag(&A2s[(l16 + 32)*132 + kt*32], hh);
    d0 = wmma_bf16(a0, bf, d0);
    d1 = wmma_bf16(a1, bf, d1);
    d2 = wmma_bf16(a2, bf, d2);
  }
  {
    float bb = b3[ncol];
    float s = 0.f;
#pragma unroll
    for (int vv = 0; vv < 8; ++vv){
      int rr = vv + 8*hh;
      s += masks[rr     ] * (d0[vv] + bb);
      s += masks[rr + 16] * (d1[vv] + bb);
      s += masks[rr + 32] * (d2[vv] + bb);
    }
    atomicAdd(&dhs[ncol], s);   // ds_add_f32; 2 lanes per column
  }
  __syncthreads();

  // ---- LN1: h_V = LN(h_V + dh/30) ----
  float x = 0.f;
  if (tid < H) x = hVrow[tid] + dhs[tid] * (1.f/30.f);
  float mean = block_sum(x, red) * (1.f/H);
  float dd = (tid < H) ? (x - mean) : 0.f;
  float var = block_sum(dd*dd, red) * (1.f/H);
  if (tid < H) hvln[tid] = dd * rsqrtf(var + 1e-5f) * g1[tid] + be1[tid];
  __syncthreads();

  // ---- FFN up: gelu(x@Win+bin), 2 hidden units per thread ----
  for (int u = tid; u < 4*H; u += 256){
    float s = bin[u];
    for (int c = 0; c < H; ++c) s += hvln[c] * Win[(size_t)c*4*H + u];
    hid[u] = gelu_f(s);
  }
  __syncthreads();

  // ---- FFN down: split-K over 512 hidden units across both thread halves ----
  float part = 0.f;
  {
    int c  = tid & 127;
    int u0 = (tid >> 7) * 256;
    for (int u = u0; u < u0 + 256; ++u) part += hid[u] * Wout[(size_t)u*H + c];
  }
  red[tid] = part;
  __syncthreads();
  float y = 0.f;
  if (tid < H) y = hvln[tid] + bout[tid] + red[tid] + red[tid + 128];
  __syncthreads();

  float mean2 = block_sum(y, red) * (1.f/H);
  float dd2 = (tid < H) ? (y - mean2) : 0.f;
  float var2 = block_sum(dd2*dd2, red) * (1.f/H);
  if (tid < H){
    float out = (dd2 * rsqrtf(var2 + 1e-5f) * g2[tid] + be2[tid]) * maskV[node];
    outV[(size_t)node*H + tid] = out;
    hVnbf[(size_t)node*H + tid] = f2bf(out);   // bf16 copy for edge kernel
  }
}

// Edge update: one block per node; async-staged [48 x 384] bf16 ->
// WMMA GEMM chain, residual + per-row LayerNorm fused, coalesced output.
__global__ __launch_bounds__(256) void edge_kernel(
  const u16* __restrict__ hVnbf, const u16* __restrict__ hEbf,
  const float* __restrict__ hE,
  const u16* __restrict__ W11t, const float* __restrict__ b11,
  const u16* __restrict__ W12t, const float* __restrict__ b12,
  const u16* __restrict__ W13t, const float* __restrict__ b13,
  const float* __restrict__ g3, const float* __restrict__ be3,
  const int* __restrict__ Eidx, float* __restrict__ outE, int Nn, int Kk)
{
  __shared__ u16   As[48*392];   // padded ld=392 (row stride 784B = 49*16B)
  __shared__ u16   A2s[48*132];
  __shared__ float X3[48*129];   // padded ld=129
  __shared__ int   idxs[48];
  __shared__ float rowm[48], rowv[48];

  const int node = blockIdx.x;
  const int bidx = node / Nn;
  const int tid  = threadIdx.x;
  const int lane = tid & 31;
  const int w    = tid >> 5;
  const int hh   = lane >> 4;
  const int l16  = lane & 15;
  const int ncol = (w << 4) + l16;

  const float* hErow   = hE   + (size_t)node * Kk * H;
  const u16*   hEbfrow = hEbf + (size_t)node * Kk * H;
  const u16*   hVnrow  = hVnbf + (size_t)node * H;

  if (tid < 48) idxs[tid] = Eidx[(size_t)node*Kk + tid];
  __syncthreads();

  // ---- async-stage A = [h_V | h_E | gather(h_V)] : 48 rows x 3 segs x 16 chunks
  for (int q = tid; q < 48*48; q += 256){
    int r   = q / 48;
    int s   = q - r*48;
    int seg = s >> 4;
    int o8  = (s & 15) << 3;
    const u16* src;
    if (seg == 0)      src = hVnrow + o8;
    else if (seg == 1) src = hEbfrow + r*H + o8;
    else               src = hVnbf + ((size_t)bidx*Nn + idxs[r])*H + o8;
    cp16(&As[r*392 + seg*128 + o8], src);
  }
  cp16_wait();
  __syncthreads();

  // ---- GEMM1: [48x384] @ [384x128] ----
  v8f acc0 = {}, acc1 = {}, acc2 = {};
  for (int kt = 0; kt < 12; ++kt){
    v16bf bf = load_frag(W11t + (size_t)ncol*384 + kt*32, hh);
    v16bf a0 = load_frag(&As[(l16     )*392 + kt*32], hh);
    v16bf a1 = load_frag(&As[(l16 + 16)*392 + kt*32], hh);
    v16bf a2 = load_frag(&As[(l16 + 32)*392 + kt*32], hh);
    acc0 = wmma_bf16(a0, bf, acc0);
    acc1 = wmma_bf16(a1, bf, acc1);
    acc2 = wmma_bf16(a2, bf, acc2);
  }
  {
    float bb = b11[ncol];
#pragma unroll
    for (int vv = 0; vv < 8; ++vv){
      int rr = vv + 8*hh;
      A2s[(rr     )*132 + ncol] = f2bf(gelu_f(acc0[vv] + bb));
      A2s[(rr + 16)*132 + ncol] = f2bf(gelu_f(acc1[vv] + bb));
      A2s[(rr + 32)*132 + ncol] = f2bf(gelu_f(acc2[vv] + bb));
    }
  }
  __syncthreads();

  // ---- GEMM2 ----
  v8f c0 = {}, c1 = {}, c2 = {};
  for (int kt = 0; kt < 4; ++kt){
    v16bf bf = load_frag(W12t + (size_t)ncol*128 + kt*32, hh);
    v16bf a0 = load_frag(&A2s[(l16     )*132 + kt*32], hh);
    v16bf a1 = load_frag(&A2s[(l16 + 16)*132 + kt*32], hh);
    v16bf a2 = load_frag(&A2s[(l16 + 32)*132 + kt*32], hh);
    c0 = wmma_bf16(a0, bf, c0);
    c1 = wmma_bf16(a1, bf, c1);
    c2 = wmma_bf16(a2, bf, c2);
  }
  __syncthreads();
  {
    float bb = b12[ncol];
#pragma unroll
    for (int vv = 0; vv < 8; ++vv){
      int rr = vv + 8*hh;
      A2s[(rr     )*132 + ncol] = f2bf(gelu_f(c0[vv] + bb));
      A2s[(rr + 16)*132 + ncol] = f2bf(gelu_f(c1[vv] + bb));
      A2s[(rr + 32)*132 + ncol] = f2bf(gelu_f(c2[vv] + bb));
    }
  }
  __syncthreads();

  // ---- GEMM3 + residual into X3 ----
  v8f d0 = {}, d1 = {}, d2 = {};
  for (int kt = 0; kt < 4; ++kt){
    v16bf bf = load_frag(W13t + (size_t)ncol*128 + kt*32, hh);
    v16bf a0 = load_frag(&A2s[(l16     )*132 + kt*32], hh);
    v16bf a1 = load_frag(&A2s[(l16 + 16)*132 + kt*32], hh);
    v16bf a2 = load_frag(&A2s[(l16 + 32)*132 + kt*32], hh);
    d0 = wmma_bf16(a0, bf, d0);
    d1 = wmma_bf16(a1, bf, d1);
    d2 = wmma_bf16(a2, bf, d2);
  }
  {
    float bb = b13[ncol];
#pragma unroll
    for (int vv = 0; vv < 8; ++vv){
      int rr = vv + 8*hh;
      X3[(rr     )*129 + ncol] = d0[vv] + bb + hErow[(rr     )*H + ncol];
      X3[(rr + 16)*129 + ncol] = d1[vv] + bb + hErow[(rr + 16)*H + ncol];
      X3[(rr + 32)*129 + ncol] = d2[vv] + bb + hErow[(rr + 32)*H + ncol];
    }
  }
  __syncthreads();

  // ---- per-row LayerNorm stats ----
  if (tid < 48){
    float s = 0.f;
    for (int c = 0; c < H; ++c) s += X3[tid*129 + c];
    float m = s * (1.f/H);
    float v = 0.f;
    for (int c = 0; c < H; ++c){ float d = X3[tid*129 + c] - m; v += d*d; }
    rowm[tid] = m; rowv[tid] = v * (1.f/H);
  }
  __syncthreads();

  // ---- normalized, coalesced output ----
  for (int e = tid; e < 48*H; e += 256){
    int r = e >> 7, c = e & 127;
    float d = X3[r*129 + c] - rowm[r];
    outE[(size_t)node*Kk*H + e] = d * rsqrtf(rowv[r] + 1e-5f) * g3[c] + be3[c];
  }
}

extern "C" void kernel_launch(void* const* d_in, const int* in_sizes, int n_in,
                              void* d_out, int out_size, void* d_ws, size_t ws_size,
                              hipStream_t stream)
{
  (void)n_in; (void)out_size; (void)ws_size;
  const float* hV   = (const float*)d_in[0];
  const float* hVa  = (const float*)d_in[1];
  const float* hE   = (const float*)d_in[2];
  const float* W1   = (const float*)d_in[3];
  const float* b1   = (const float*)d_in[4];
  const float* W2   = (const float*)d_in[5];
  const float* b2   = (const float*)d_in[6];
  const float* W3   = (const float*)d_in[7];
  const float* b3   = (const float*)d_in[8];
  const float* W11  = (const float*)d_in[9];
  const float* b11  = (const float*)d_in[10];
  const float* W12  = (const float*)d_in[11];
  const float* b12  = (const float*)d_in[12];
  const float* W13  = (const float*)d_in[13];
  const float* b13  = (const float*)d_in[14];
  const float* Win  = (const float*)d_in[15];
  const float* bin  = (const float*)d_in[16];
  const float* Wout = (const float*)d_in[17];
  const float* bout = (const float*)d_in[18];
  const float* g1   = (const float*)d_in[19];
  const float* be1  = (const float*)d_in[20];
  const float* g2   = (const float*)d_in[21];
  const float* be2  = (const float*)d_in[22];
  const float* g3   = (const float*)d_in[23];
  const float* be3  = (const float*)d_in[24];
  const float* mV   = (const float*)d_in[25];
  const float* mA   = (const float*)d_in[26];
  const int*   Eix  = (const int*)d_in[27];

  const int BN = in_sizes[0] / H;    // B*N = 4096
  const int Kk = in_sizes[27] / BN;  // 48
  const int B  = 2;                  // per reference setup
  const int Nn = BN / B;             // 2048

  // Workspace layout (u16 units): transposed bf16 weights, then bf16 tensors.
  u16* W1t   = (u16*)d_ws;                 // 128*640
  u16* W2t   = W1t  + 128*640;             // 128*128
  u16* W3t   = W2t  + 128*128;
  u16* W11t  = W3t  + 128*128;             // 128*384
  u16* W12t  = W11t + 128*384;
  u16* W13t  = W12t + 128*128;
  u16* hEbf  = W13t + 128*128;             // BN*Kk*H
  u16* hVbf  = hEbf + (size_t)BN*Kk*H;     // BN*H
  u16* hVabf = hVbf + (size_t)BN*H;
  u16* hVnbf = hVabf + (size_t)BN*H;       // updated h_V, bf16

  wconv_kernel<<<(128*640+255)/256, 256, 0, stream>>>(W1,  W1t,  640);
  wconv_kernel<<<(128*128+255)/256, 256, 0, stream>>>(W2,  W2t,  128);
  wconv_kernel<<<(128*128+255)/256, 256, 0, stream>>>(W3,  W3t,  128);
  wconv_kernel<<<(128*384+255)/256, 256, 0, stream>>>(W11, W11t, 384);
  wconv_kernel<<<(128*128+255)/256, 256, 0, stream>>>(W12, W12t, 128);
  wconv_kernel<<<(128*128+255)/256, 256, 0, stream>>>(W13, W13t, 128);

  {
    int nE4 = (BN * Kk * H) / 4, nV4 = (BN * H) / 4;
    bconv_kernel<<<(nE4+255)/256, 256, 0, stream>>>(hE,  hEbf,  nE4);
    bconv_kernel<<<(nV4+255)/256, 256, 0, stream>>>(hV,  hVbf,  nV4);
    bconv_kernel<<<(nV4+255)/256, 256, 0, stream>>>(hVa, hVabf, nV4);
  }

  float* outV = (float*)d_out;
  float* outE = outV + (size_t)BN * H;

  node_kernel<<<BN, 256, 0, stream>>>(hV, hVbf, hVabf, hEbf,
                                      W1t, b1, W2t, b2, W3t, b3,
                                      Win, bin, Wout, bout, g1, be1, g2, be2,
                                      mV, mA, Eix, outV, hVnbf, Nn, Kk);
  edge_kernel<<<BN, 256, 0, stream>>>(hVnbf, hEbf, hE,
                                      W11t, b11, W12t, b12, W13t, b13,
                                      g3, be3, Eix, outE, Nn, Kk);
}
